// GCNBMPEncoder_23811298690073
// MI455X (gfx1250) — matrix operand order repaired and checked by reading, not compiled, not run on hardware
//
#include <hip/hip_runtime.h>
#include <hip/hip_bf16.h>

#define FD 64
#define FR 4

typedef __attribute__((ext_vector_type(16))) _Float16 v16h;
typedef __attribute__((ext_vector_type(8)))  float    v8f;

// ---------------------------------------------------------------------------
// WMMA A-fragment swizzle for 16-bit A 16x32 (ISA 7.12.2):
//  lane = m + 16*h where h selects K-halves {0-7 / 8-15} and {16-23 / 24-31}
//  VGPR v: v0..3 -> K = h*8 + 2v + p (K<16), v4..7 -> 16 + h*8 + 2(v-4) + p
// Returns half-index (lane*16 + elem) for tile element (m, k), k in [0,32).
// ---------------------------------------------------------------------------
__device__ __forceinline__ int swizzleA(int m, int k) {
    int hi  = k >> 4;          // K >= 16 -> upper VGPR quad
    int k2  = k & 15;
    int h   = k2 >> 3;         // lane half
    int rem = k2 & 7;
    int v   = (hi << 2) + (rem >> 1);
    int p   = rem & 1;
    int lane = m + (h << 4);
    int e    = (v << 1) + p;
    return lane * 16 + e;
}

// ---------------------------------------------------------------------------
// Pack conv weights: Bconv[k][j] = (k<256 ? w[j][k] : ws[j][k-256]) as f16,
// stored in B-fragment order: packed[((kt*4+jt)*32+lane)*16+e],
// where k = kt*32 + (e>>3)*16 + (lane>>4)*8 + (e&7), j = jt*16 + (lane&15).
// ---------------------------------------------------------------------------
__global__ void pack_conv_kernel(const float* __restrict__ w,
                                 const float* __restrict__ ws,
                                 _Float16* __restrict__ out) {
    int tid = blockIdx.x * blockDim.x + threadIdx.x;
    if (tid >= 10 * 4 * 32 * 16) return;
    int e    = tid & 15;
    int lane = (tid >> 4) & 31;
    int jt   = (tid >> 9) & 3;
    int kt   = tid >> 11;
    int k = kt * 32 + ((e >> 3) << 4) + ((lane >> 4) << 3) + (e & 7);
    int j = jt * 16 + (lane & 15);
    float v = (k < 256) ? w[j * 256 + k] : ws[j * 64 + (k - 256)];
    out[tid] = (_Float16)v;
}

// Same layout for a 128x64 highway weight (pw.T or tw.T), pwt[k][j]=pw[j*128+k]
__global__ void pack_hw_kernel(const float* __restrict__ pw,
                               _Float16* __restrict__ out) {
    int tid = blockIdx.x * blockDim.x + threadIdx.x;
    if (tid >= 4 * 4 * 32 * 16) return;
    int e    = tid & 15;
    int lane = (tid >> 4) & 31;
    int jt   = (tid >> 9) & 3;
    int kt   = tid >> 11;
    int k = kt * 32 + ((e >> 3) << 4) + ((lane >> 4) << 3) + (e & 7);
    int j = jt * 16 + (lane & 15);
    out[tid] = (_Float16)pw[j * 128 + k];
}

// ---------------------------------------------------------------------------
// Edge scatter: acc[(dst*R+rel)*64 + d] += x[src*64 + d]; cnt[seg] += 1.
// One thread per (edge, feature). acc (102 MB) + x (25 MB) are L2-resident.
// ---------------------------------------------------------------------------
__global__ void scatter_kernel(const float* __restrict__ x,
                               const int* __restrict__ src,
                               const int* __restrict__ dst,
                               const int* __restrict__ rel,
                               float* __restrict__ acc,
                               float* __restrict__ cnt,
                               int nEdges) {
    unsigned long long gid =
        (unsigned long long)blockIdx.x * blockDim.x + threadIdx.x;
    if (gid >= (unsigned long long)nEdges * FD) return;
    int e = (int)(gid >> 6);
    int d = (int)(gid & 63);
    int s   = src[e];
    int seg = dst[e] * FR + rel[e];
    atomicAdd(acc + (size_t)seg * FD + d, x[(size_t)s * FD + d]);
    if (d == 0) atomicAdd(cnt + seg, 1.0f);
}

// ---------------------------------------------------------------------------
// Fused per-layer node kernel. One block = 16 node rows, 4 wave32s; each wave
// owns a 16-column strip of the 64 output columns.
//   h   = sigmoid([mean(acc) | xin] @ Bconv + (b+bs))          (K=320, 10 WMMA)
//   pr  = relu   ([h | prev] @ Bp + pb)                        (K=128, 4 WMMA)
//   g   = sigmoid([h | prev] @ Bt + tb)                        (K=128, 4 WMMA)
//   out = g*pr + (1-g)*h
// ---------------------------------------------------------------------------
__global__ __launch_bounds__(128) void rgc_highway_kernel(
    const float* __restrict__ acc, const float* __restrict__ cnt,
    const float* __restrict__ xin, const float* __restrict__ prev,
    const _Float16* __restrict__ Bconv,
    const float* __restrict__ b, const float* __restrict__ bs,
    const _Float16* __restrict__ Bp, const float* __restrict__ pb,
    const _Float16* __restrict__ Bt, const float* __restrict__ tb,
    float* __restrict__ hOut, float* __restrict__ gOut, int Nn) {

    __shared__ __align__(32) _Float16 aFrag[10 * 512];  // 16x320 A fragments
    __shared__ __align__(32) _Float16 a2Frag[4 * 512];  // 16x128 A fragments
    __shared__ float hTile[16 * 64];
    __shared__ float pTile[16 * 64];

    const int tid  = threadIdx.x;
    const int wave = tid >> 5;
    const int lane = tid & 31;
    const int m0   = blockIdx.x * 16;

    // Stage prev tile (highway carry input)
    for (int i = tid; i < 16 * 64; i += 128) {
        int m = i >> 6, row = m0 + m;
        pTile[i] = (row < Nn) ? prev[(size_t)row * 64 + (i & 63)] : 0.0f;
    }
    // Stage A = [scatter-mean(256) | xin(64)] into WMMA fragment layout (f16)
    for (int idx = tid; idx < 16 * 320; idx += 128) {
        int m = idx / 320;
        int k = idx - m * 320;
        int row = m0 + m;
        float v = 0.0f;
        if (row < Nn) {
            if (k < 256) {
                float c = cnt[(size_t)row * FR + (k >> 6)];
                v = acc[(size_t)row * 256 + k] / fmaxf(c, 1.0f);
            } else {
                v = xin[(size_t)row * 64 + (k - 256)];
            }
        }
        aFrag[(k >> 5) * 512 + swizzleA(m, k & 31)] = (_Float16)v;
    }
    __syncthreads();

    const int n    = lane & 15;
    const int j    = wave * 16 + n;       // output column for this lane
    const int mrow = (lane >> 4) << 3;    // C/D: VGPR r holds M = r + 8*(lane/16)

    // ---- RGC GEMM: 10 x v_wmma_f32_16x16x32_f16 over K=320 ----
    v8f cacc;
    {
        float bias = b[j] + bs[j];
        #pragma unroll
        for (int i = 0; i < 8; i++) cacc[i] = bias;
    }
    #pragma unroll
    for (int kt = 0; kt < 10; kt++) {
        v16h a  = *(const v16h*)(aFrag + kt * 512 + lane * 16);
        v16h bm = *(const v16h*)(Bconv + (((kt * 4 + wave) * 32 + lane) << 4));
        cacc = __builtin_amdgcn_wmma_f32_16x16x32_f16(
            false, a, false, bm, (short)0, cacc, false, false);
    }
    #pragma unroll
    for (int i = 0; i < 8; i++) {
        int M = i + mrow;
        float hv = 1.0f / (1.0f + __expf(-cacc[i]));
        hTile[M * 64 + j] = hv;
        int row = m0 + M;
        if (row < Nn) hOut[(size_t)row * 64 + j] = hv;
    }
    __syncthreads();

    // Stage A2 = [h | prev] into fragment layout
    for (int idx = tid; idx < 16 * 128; idx += 128) {
        int m = idx >> 7;
        int k = idx & 127;
        float v = (k < 64) ? hTile[m * 64 + k] : pTile[m * 64 + (k - 64)];
        a2Frag[(k >> 5) * 512 + swizzleA(m, k & 31)] = (_Float16)v;
    }
    __syncthreads();

    // ---- Highway: two GEMMs sharing A2, 4+4 WMMAs ----
    v8f pacc, gacc;
    {
        float pbv = pb[j], tbv = tb[j];
        #pragma unroll
        for (int i = 0; i < 8; i++) { pacc[i] = pbv; gacc[i] = tbv; }
    }
    #pragma unroll
    for (int kt = 0; kt < 4; kt++) {
        v16h a   = *(const v16h*)(a2Frag + kt * 512 + lane * 16);
        v16h bpm = *(const v16h*)(Bp + (((kt * 4 + wave) * 32 + lane) << 4));
        v16h btm = *(const v16h*)(Bt + (((kt * 4 + wave) * 32 + lane) << 4));
        pacc = __builtin_amdgcn_wmma_f32_16x16x32_f16(
            false, a, false, bpm, (short)0, pacc, false, false);
        gacc = __builtin_amdgcn_wmma_f32_16x16x32_f16(
            false, a, false, btm, (short)0, gacc, false, false);
    }
    #pragma unroll
    for (int i = 0; i < 8; i++) {
        int M = i + mrow;
        int row = m0 + M;
        if (row < Nn) {
            float pr = fmaxf(pacc[i], 0.0f);
            float gv = 1.0f / (1.0f + __expf(-gacc[i]));
            float hv = hTile[M * 64 + j];
            gOut[(size_t)row * 64 + j] = gv * pr + (1.0f - gv) * hv;
        }
    }
}

// ---------------------------------------------------------------------------
extern "C" void kernel_launch(void* const* d_in, const int* in_sizes, int n_in,
                              void* d_out, int out_size, void* d_ws, size_t ws_size,
                              hipStream_t stream) {
    const float* x   = (const float*)d_in[0];
    const int*   src = (const int*)d_in[1];
    const int*   dst = (const int*)d_in[2];
    const int*   rel = (const int*)d_in[3];

    const float* c1w  = (const float*)d_in[4];
    const float* c1b  = (const float*)d_in[5];
    const float* c1ws = (const float*)d_in[6];
    const float* c1bs = (const float*)d_in[7];
    const float* h1pw = (const float*)d_in[8];
    const float* h1pb = (const float*)d_in[9];
    const float* h1tw = (const float*)d_in[10];
    const float* h1tb = (const float*)d_in[11];
    const float* c2w  = (const float*)d_in[12];
    const float* c2b  = (const float*)d_in[13];
    const float* c2ws = (const float*)d_in[14];
    const float* c2bs = (const float*)d_in[15];
    const float* h2pw = (const float*)d_in[16];
    const float* h2pb = (const float*)d_in[17];
    const float* h2tw = (const float*)d_in[18];
    const float* h2tb = (const float*)d_in[19];

    const int N_ = in_sizes[0] / FD;
    const int E_ = in_sizes[1];

    // Workspace carve-up (256B aligned slices)
    char* p = (char*)d_ws;
    auto take = [&](size_t bytes) {
        char* r = p;
        p += (bytes + 255) & ~(size_t)255;
        return r;
    };
    float*    acc    = (float*)take((size_t)N_ * FR * FD * sizeof(float));
    float*    cnt    = (float*)take((size_t)N_ * FR * sizeof(float));
    float*    h1     = (float*)take((size_t)N_ * FD * sizeof(float));
    float*    g1     = (float*)take((size_t)N_ * FD * sizeof(float));
    float*    h2     = (float*)take((size_t)N_ * FD * sizeof(float));
    _Float16* Bconv1 = (_Float16*)take(20480 * sizeof(_Float16));
    _Float16* Bp1    = (_Float16*)take(8192 * sizeof(_Float16));
    _Float16* Bt1    = (_Float16*)take(8192 * sizeof(_Float16));
    _Float16* Bconv2 = (_Float16*)take(20480 * sizeof(_Float16));
    _Float16* Bp2    = (_Float16*)take(8192 * sizeof(_Float16));
    _Float16* Bt2    = (_Float16*)take(8192 * sizeof(_Float16));
    float* out = (float*)d_out;

    // Weight pre-swizzle (tiny, L2-resident afterwards)
    pack_conv_kernel<<<80, 256, 0, stream>>>(c1w, c1ws, Bconv1);
    pack_hw_kernel<<<32, 256, 0, stream>>>(h1pw, Bp1);
    pack_hw_kernel<<<32, 256, 0, stream>>>(h1tw, Bt1);
    pack_conv_kernel<<<80, 256, 0, stream>>>(c2w, c2ws, Bconv2);
    pack_hw_kernel<<<32, 256, 0, stream>>>(h2pw, Bp2);
    pack_hw_kernel<<<32, 256, 0, stream>>>(h2tw, Bt2);

    const unsigned long long edgeWork = (unsigned long long)E_ * FD;
    const int edgeBlocks = (int)((edgeWork + 255) / 256);
    const int nodeBlocks = (N_ + 15) / 16;

    // ---- Layer 1 ----
    hipMemsetAsync(acc, 0, (size_t)N_ * FR * FD * sizeof(float), stream);
    hipMemsetAsync(cnt, 0, (size_t)N_ * FR * sizeof(float), stream);
    scatter_kernel<<<edgeBlocks, 256, 0, stream>>>(x, src, dst, rel, acc, cnt, E_);
    rgc_highway_kernel<<<nodeBlocks, 128, 0, stream>>>(
        acc, cnt, x, /*prev=*/x, Bconv1, c1b, c1bs, Bp1, h1pb, Bt1, h1tb,
        h1, g1, N_);

    // ---- Layer 2 ----
    hipMemsetAsync(acc, 0, (size_t)N_ * FR * FD * sizeof(float), stream);
    hipMemsetAsync(cnt, 0, (size_t)N_ * FR * sizeof(float), stream);
    scatter_kernel<<<edgeBlocks, 256, 0, stream>>>(g1, src, dst, rel, acc, cnt, E_);
    rgc_highway_kernel<<<nodeBlocks, 128, 0, stream>>>(
        acc, cnt, g1, /*prev=*/h1, Bconv2, c2b, c2bs, Bp2, h2pb, Bt2, h2tb,
        h2, out, N_);
}
